// MultiHeadAttention_39049842655283
// MI455X (gfx1250) — matrix operand order repaired and checked by reading, not compiled, not run on hardware
//
#include <hip/hip_runtime.h>

#define NUM_HEADS 16
#define N_EMBED   1024
#define HEAD_SIZE 64
#define SEQ_T     1024
#define BATCH     8

typedef __attribute__((ext_vector_type(16))) __bf16 v16bf;
typedef __attribute__((ext_vector_type(8)))  float  v8f;
typedef unsigned int u32x4 __attribute__((ext_vector_type(4)));
typedef int          i32x4v __attribute__((ext_vector_type(4)));
typedef int          i32x8v __attribute__((ext_vector_type(8)));

#if defined(__has_builtin)
#define HAS_TDM        __has_builtin(__builtin_amdgcn_tensor_load_to_lds)
#define HAS_WAIT_TENS  __has_builtin(__builtin_amdgcn_s_wait_tensorcnt)
#define HAS_CVTPK      __has_builtin(__builtin_amdgcn_cvt_pk_bf16_f32)
#else
#define HAS_TDM 0
#define HAS_WAIT_TENS 0
#define HAS_CVTPK 0
#endif

union V16U { unsigned int u[8]; unsigned short s[16]; v16bf v; };
union V16Q { uint4 q[2]; v16bf v; };

static __device__ inline unsigned short f2bf(float f) {
  unsigned int u = __float_as_uint(f);
  u += 0x7FFFu + ((u >> 16) & 1u);          // round-to-nearest-even
  return (unsigned short)(u >> 16);
}

// pack two f32 -> packed bf16x2 (v_cvt_pk_bf16_f32 when available)
static __device__ inline unsigned int pack2bf(float a, float b) {
#if HAS_CVTPK
  auto p = __builtin_amdgcn_cvt_pk_bf16_f32(a, b);
  unsigned int u;
  __builtin_memcpy(&u, &p, 4);
  return u;
#else
  return (unsigned int)f2bf(a) | ((unsigned int)f2bf(b) << 16);
#endif
}

static __device__ inline v8f zero8() {
  v8f z;
#pragma unroll
  for (int i = 0; i < 8; ++i) z[i] = 0.0f;
  return z;
}

// A fragment, 16x32 bf16 (M x K). lds row-major [m][k], k-contiguous.
// lane L: m=L%16, hi=8 for upper half-wave; halves are contiguous 16B runs:
// elems 0..7  -> k = hi..hi+7        (base + hi)
// elems 8..15 -> k = 16+hi..16+hi+7  (base + 16 + hi)
static __device__ inline v16bf load_a16(const unsigned short* lds, int mbase,
                                        int stride, int kbase, int lane) {
  const int m  = mbase + (lane & 15);
  const int hi = (lane >> 4) ? 8 : 0;
  const unsigned short* base = lds + m * stride + kbase + hi;   // 16B aligned
  V16Q r;
  r.q[0] = *(const uint4*)base;
  r.q[1] = *(const uint4*)(base + 16);
  return r.v;
}

// B fragment, 32x16 bf16 (K x N). lds is [n][k], k-contiguous.
// lane L: n=L%16, hi=8 for upper half-wave; elems 0..7 -> k = hi+i, elems 8..15 -> k = 16+hi+(i-8).
static __device__ inline v16bf load_b16(const unsigned short* lds, int nbase,
                                        int stride, int kbase, int lane) {
  const int n  = nbase + (lane & 15);
  const int hi = (lane >> 4) ? 8 : 0;
  const unsigned short* base = lds + n * stride + kbase + hi;   // 16B aligned
  V16Q r;
  r.q[0] = *(const uint4*)base;
  r.q[1] = *(const uint4*)(base + 16);
  return r.v;
}

// A fragment assembled from f32 LDS strip (softmax probs), packed-converted.
static __device__ inline v16bf load_a32f(const float* lds, int mbase,
                                         int stride, int kbase, int lane) {
  const int m   = mbase + (lane & 15);
  const int hi8 = (lane >> 4) ? 8 : 0;
  const float* row = lds + m * stride + kbase;
  V16U r;
#pragma unroll
  for (int j = 0; j < 4; ++j)
    r.u[j] = pack2bf(row[hi8 + 2 * j], row[hi8 + 2 * j + 1]);
#pragma unroll
  for (int j = 0; j < 4; ++j)
    r.u[4 + j] = pack2bf(row[16 + hi8 + 2 * j], row[16 + hi8 + 2 * j + 1]);
  return r.v;
}

static __device__ inline v8f wmma_bf16(v16bf a, v16bf b, v8f c) {
  return __builtin_amdgcn_wmma_f32_16x16x32_bf16(false, a, false, b,
                                                 (short)0, c, false, false);
}

#if HAS_TDM
// Issue a TDM 2D (or 1D when tile1==0) bf16 tile load global->LDS.
// Descriptor layout per CDNA5 ISA 8.3/8.4 (group0 128b, group1 256b).
// This toolchain exposes the 6-arg builtin: (g0, g1, g2, g3, g_extra, cpol).
static __device__ inline void tdm_load_2d(unsigned lds_addr, unsigned long long gaddr,
                                          unsigned dim0, unsigned dim1,
                                          unsigned tile0, unsigned tile1,
                                          unsigned long long stride0) {
  u32x4 g0;
  g0[0] = 1u;                                              // count=1, user D#
  g0[1] = lds_addr;                                        // lds_addr[31:0]
  g0[2] = (unsigned)(gaddr & 0xFFFFFFFFull);               // global_addr[31:0]
  g0[3] = (unsigned)((gaddr >> 32) & 0x1FFFFFFull) | (2u << 30);  // addr[56:32]|type=2
  i32x8v g1;
  g1[0] = (int)(1u << 16);                                 // data_size=1 (2B)
  g1[1] = (int)((dim0 & 0xFFFFu) << 16);                   // tensor_dim0 lo
  g1[2] = (int)((dim0 >> 16) | ((dim1 & 0xFFFFu) << 16));  // dim0 hi | dim1 lo
  g1[3] = (int)((dim1 >> 16) | (tile0 << 16));             // dim1 hi | tile_dim0
  g1[4] = (int)tile1;                                      // tile_dim1 (tile_dim2=0)
  g1[5] = (int)(stride0 & 0xFFFFFFFFull);                  // tensor_dim0_stride lo
  g1[6] = (int)((stride0 >> 32) & 0xFFFFull);              // stride hi (dim1_stride=0)
  g1[7] = 0;
  i32x4v z4 = {0, 0, 0, 0};
  i32x8v z8 = {0, 0, 0, 0, 0, 0, 0, 0};
  __builtin_amdgcn_tensor_load_to_lds(g0, g1, z4, z4, z8, 0);
}

static __device__ inline unsigned lds_addr_of(const void* p) {
  return (unsigned)(unsigned long long)(const __attribute__((address_space(3))) void*)p;
}
#endif

static __device__ inline void wait_tensor0() {
#if HAS_WAIT_TENS
  __builtin_amdgcn_s_wait_tensorcnt(0);
#else
  asm volatile("s_wait_tensorcnt 0x0" ::: "memory");
#endif
}

// ---------------------------------------------------------------------------
// Kernel 1: QKV projection. One block = (b, 64-row tile of T, head h).
// Q,K stored [t,d] bf16, V stored transposed [d,t] bf16.
// ---------------------------------------------------------------------------
__global__ __launch_bounds__(256) void qkv_proj_kernel(
    const float* __restrict__ x, const float* __restrict__ Wq,
    const float* __restrict__ Wk, const float* __restrict__ Wv,
    unsigned short* __restrict__ qo, unsigned short* __restrict__ ko,
    unsigned short* __restrict__ vto) {
  __shared__ unsigned int lxu[64 * 32];       // X tile  [t][c] packed bf16x2
  __shared__ unsigned int lwu[3 * 64 * 32];   // W tiles [mat][d][c] packed bf16x2
  const unsigned short* lxs = (const unsigned short*)lxu;
  const unsigned short* lws = (const unsigned short*)lwu;

  const int tid  = threadIdx.x;
  const int lane = tid & 31;
  const int w    = tid >> 5;
  const int bx   = blockIdx.x;
  const int h    = bx & 15;
  const int tt   = (bx >> 4) & 15;
  const int b    = bx >> 8;
  const int tbase = tt * 64;
  const int bh    = b * NUM_HEADS + h;

  const float* Ws[3] = {Wq + (size_t)h * N_EMBED * HEAD_SIZE,
                        Wk + (size_t)h * N_EMBED * HEAD_SIZE,
                        Wv + (size_t)h * N_EMBED * HEAD_SIZE};

  const int r      = w & 3;    // row tile (16 rows)
  const int cgroup = w >> 2;   // 0: col tiles 0..5, 1: col tiles 6..11

  v8f acc[6];
#pragma unroll
  for (int j = 0; j < 6; ++j) acc[j] = zero8();

  for (int cb = 0; cb < N_EMBED / 64; ++cb) {
    const int cbase = cb * 64;
#pragma unroll
    for (int i = 0; i < 8; ++i) {               // X: 2048 bf16x2 pairs
      const int e = tid + 256 * i;
      const int row = e >> 5, c2 = e & 31;
      const float2 v = ((const float2*)(x + ((size_t)(b * SEQ_T + tbase + row)) *
                                                N_EMBED + cbase))[c2];
      lxu[row * 32 + c2] = pack2bf(v.x, v.y);
    }
#pragma unroll
    for (int m3 = 0; m3 < 3; ++m3) {            // W: 3 x 2048 pairs (coalesced in d)
      const float* W = Ws[m3];
#pragma unroll
      for (int i = 0; i < 8; ++i) {
        const int e = tid + 256 * i;
        const int d = e & 63, c2 = e >> 6;
        const float w0 = W[(size_t)(cbase + 2 * c2) * HEAD_SIZE + d];
        const float w1 = W[(size_t)(cbase + 2 * c2 + 1) * HEAD_SIZE + d];
        lwu[m3 * 2048 + d * 32 + c2] = pack2bf(w0, w1);
      }
    }
    if (cb + 1 < N_EMBED / 64) {                // prefetch next X chunk
      const int row = tid >> 2, seg = tid & 3;
      __builtin_prefetch(x + ((size_t)(b * SEQ_T + tbase + row)) * N_EMBED +
                             cbase + 64 + seg * 16, 0, 0);
    }
    __syncthreads();

#pragma unroll
    for (int ks = 0; ks < 2; ++ks) {
      const v16bf a = load_a16(lxs, r * 16, 64, ks * 32, lane);
      v16bf bfr[6];
#pragma unroll
      for (int j = 0; j < 6; ++j) {             // batch all B-frag LDS loads
        const int ct = cgroup * 6 + j;
        const int mat = ct >> 2, dt = ct & 3;
        bfr[j] = load_b16(lws + mat * 4096, dt * 16, 64, ks * 32, lane);
      }
#pragma unroll
      for (int j = 0; j < 6; ++j)               // then back-to-back WMMAs
        acc[j] = wmma_bf16(a, bfr[j], acc[j]);
    }
    __syncthreads();
  }

  const int n   = lane & 15;
  const int hi8 = (lane >> 4) ? 8 : 0;
#pragma unroll
  for (int j = 0; j < 6; ++j) {
    const int ct = cgroup * 6 + j;
    const int mat = ct >> 2, dt = ct & 3;
#pragma unroll
    for (int rr = 0; rr < 8; ++rr) {
      const int t = tbase + r * 16 + rr + hi8;
      const int d = dt * 16 + n;
      const unsigned short bits = f2bf(acc[j][rr]);
      if (mat == 0)      qo[((size_t)bh * SEQ_T + t) * HEAD_SIZE + d] = bits;
      else if (mat == 1) ko[((size_t)bh * SEQ_T + t) * HEAD_SIZE + d] = bits;
      else               vto[((size_t)bh * HEAD_SIZE + d) * SEQ_T + t] = bits;
    }
  }
}

// ---------------------------------------------------------------------------
// Kernel 2: causal attention. One block = (b, h, 32 query rows).
// Full [32 x 1024] f32 score strip in LDS; K/V/Q staged via TDM.
// ---------------------------------------------------------------------------
#define ATTN_SMEM_BYTES (131072 + 32768 + 4096 + 1024 + 128 + 128)

__global__ __launch_bounds__(256) void attn_kernel(
    const unsigned short* __restrict__ qo, const unsigned short* __restrict__ ko,
    const unsigned short* __restrict__ vto, float* __restrict__ out) {
  extern __shared__ char smem[];
  float*          sS   = (float*)smem;                          // [32][1024] f32
  unsigned short* sKV  = (unsigned short*)(smem + 131072);      // K:[s][d] / V:[d][s]
  unsigned short* sQ   = (unsigned short*)(smem + 131072 + 32768);
  float*          sRed = (float*)(smem + 131072 + 32768 + 4096);
  float*          sMax = (float*)(smem + 131072 + 32768 + 4096 + 1024);
  float*          sSum = (float*)(smem + 131072 + 32768 + 4096 + 1024 + 128);

  const int tid  = threadIdx.x;
  const int lane = tid & 31;
  const int w    = tid >> 5;
  const int bx   = blockIdx.x;
  const int h    = bx & 15;
  const int qt   = (bx >> 4) & 31;
  const int b    = bx >> 9;
  const int bh   = b * NUM_HEADS + h;
  const int qbase = qt * 32;
  const int kvlen = qbase + 32;       // causal: only keys s <= qbase+31 matter

  // ---- Q tile [32][64] bf16 (contiguous 4KB)
#if HAS_TDM
  if (w == 0) {
    tdm_load_2d(lds_addr_of(sQ),
                (unsigned long long)(size_t)(qo + ((size_t)bh * SEQ_T + qbase) * HEAD_SIZE),
                2048, 1, 2048, 0, 2048);
    wait_tensor0();
  }
#else
  {
    const uint4* qg = (const uint4*)(qo + ((size_t)bh * SEQ_T + qbase) * HEAD_SIZE);
    uint4* qs = (uint4*)sQ;
#pragma unroll
    for (int i = 0; i < 2; ++i) qs[tid + 256 * i] = qg[tid + 256 * i];
  }
#endif
  __syncthreads();

  const int r    = w & 1;      // query row tile (16 rows)
  const int cpar = w >> 1;     // 0..3
  const v16bf aq0 = load_a16(sQ, r * 16, 64, 0, lane);
  const v16bf aq1 = load_a16(sQ, r * 16, 64, 32, lane);

  // ---- scores: stream K chunks of 256 keys (always full chunk; tail unused)
  const int nch = (kvlen + 255) >> 8;
  for (int ci = 0; ci < nch; ++ci) {
    const int c0 = ci * 256;
    const int clen = (kvlen - c0 < 256) ? (kvlen - c0) : 256;
#if HAS_TDM
    if (w == 0) {   // contiguous 32KB: [256 keys][64 d] bf16
      tdm_load_2d(lds_addr_of(sKV),
                  (unsigned long long)(size_t)(ko + ((size_t)bh * SEQ_T + c0) * HEAD_SIZE),
                  16384, 1, 16384, 0, 16384);
      wait_tensor0();
    }
#else
    {
      const uint4* kg = (const uint4*)(ko + ((size_t)bh * SEQ_T + c0) * HEAD_SIZE);
      uint4* kd = (uint4*)sKV;
#pragma unroll
      for (int i = 0; i < 8; ++i) kd[tid + 256 * i] = kg[tid + 256 * i];
    }
#endif
    __syncthreads();

    const int nct = clen >> 4;
    for (int ct = cpar; ct < nct; ct += 4) {
      v8f acc = zero8();
      const v16bf b0 = load_b16(sKV, ct * 16, 64, 0, lane);
      const v16bf b1 = load_b16(sKV, ct * 16, 64, 32, lane);
      acc = wmma_bf16(aq0, b0, acc);
      acc = wmma_bf16(aq1, b1, acc);
      const int n = lane & 15;
      const int hi8 = (lane >> 4) ? 8 : 0;
      const int col = c0 + ct * 16 + n;
#pragma unroll
      for (int rr = 0; rr < 8; ++rr)
        sS[(r * 16 + rr + hi8) * SEQ_T + col] = acc[rr] * 0.125f;  // hs^-0.5
    }
    __syncthreads();
  }

  // ---- causal softmax (8 partials x 128 cols per row)
  {
    const int lr = tid & 31, part = tid >> 5;
    const int count = qbase + lr + 1;
    const int j0 = part * 128;
    const int j1 = (j0 + 128 < kvlen) ? j0 + 128 : kvlen;
    float pm = -3.4e38f;
    for (int j = j0; j < j1; ++j)
      if (j < count) pm = fmaxf(pm, sS[lr * SEQ_T + j]);
    sRed[part * 32 + lr] = pm;
    __syncthreads();
    if (tid < 32) {
      float m = -3.4e38f;
#pragma unroll
      for (int p = 0; p < 8; ++p) m = fmaxf(m, sRed[p * 32 + tid]);
      sMax[tid] = m;
    }
    __syncthreads();
    const float rm = sMax[lr];
    float ps = 0.0f;
    for (int j = j0; j < j1; ++j) {
      float v = 0.0f;
      if (j < count) { v = __expf(sS[lr * SEQ_T + j] - rm); ps += v; }
      sS[lr * SEQ_T + j] = v;
    }
    sRed[part * 32 + lr] = ps;
    __syncthreads();
    if (tid < 32) {
      float s = 0.0f;
#pragma unroll
      for (int p = 0; p < 8; ++p) s += sRed[p * 32 + tid];
      sSum[tid] = s;
    }
    __syncthreads();
  }

  // ---- P @ V: stream V^T chunks into LDS [d][256]
  const int ctv = cpar;
  v8f oacc = zero8();
  for (int ci = 0; ci < nch; ++ci) {
    const int c0 = ci * 256;
    const int clen = (kvlen - c0 < 256) ? (kvlen - c0) : 256;
#if HAS_TDM
    if (w == 0) {   // 2D tile: 64 rows (d) x 256 keys, row stride T
      tdm_load_2d(lds_addr_of(sKV),
                  (unsigned long long)(size_t)(vto + ((size_t)bh * HEAD_SIZE) * SEQ_T + c0),
                  SEQ_T, HEAD_SIZE, 256, HEAD_SIZE, SEQ_T);
      wait_tensor0();
    }
#else
#pragma unroll
    for (int i = 0; i < 8; ++i) {
      const int e = tid + 256 * i;
      const int row = e >> 5, c4 = e & 31;
      ((uint4*)(sKV + row * 256))[c4] =
          ((const uint4*)(vto + ((size_t)bh * HEAD_SIZE + row) * SEQ_T + c0))[c4];
    }
#endif
    __syncthreads();
    const int nks = clen >> 5;
    for (int ks = 0; ks < nks; ++ks) {
      const v16bf ap = load_a32f(sS, r * 16, SEQ_T, c0 + ks * 32, lane);
      const v16bf bv = load_b16(sKV, ctv * 16, 256, ks * 32, lane);
      oacc = wmma_bf16(ap, bv, oacc);
    }
    __syncthreads();
  }

  // ---- out[b, t, h*64 + d] = oacc / rowsum
  const int n = lane & 15;
  const int hi8 = (lane >> 4) ? 8 : 0;
#pragma unroll
  for (int rr = 0; rr < 8; ++rr) {
    const int lr = r * 16 + rr + hi8;
    const int t = qbase + lr;
    const int d = ctv * 16 + n;
    out[((size_t)(b * SEQ_T + t)) * (NUM_HEADS * HEAD_SIZE) + h * HEAD_SIZE + d] =
        oacc[rr] / sSum[lr];
  }
}

// ---------------------------------------------------------------------------
extern "C" void kernel_launch(void* const* d_in, const int* in_sizes, int n_in,
                              void* d_out, int out_size, void* d_ws, size_t ws_size,
                              hipStream_t stream) {
  const float* x  = (const float*)d_in[0];
  const float* Wq = (const float*)d_in[1];
  const float* Wk = (const float*)d_in[2];
  const float* Wv = (const float*)d_in[3];
  float* out = (float*)d_out;

  const size_t QSZ = (size_t)BATCH * NUM_HEADS * SEQ_T * HEAD_SIZE;  // bf16 elems
  unsigned short* wsq  = (unsigned short*)d_ws;
  unsigned short* wsk  = wsq + QSZ;
  unsigned short* wsvt = wsk + QSZ;

  qkv_proj_kernel<<<BATCH * (SEQ_T / 64) * NUM_HEADS, 256, 0, stream>>>(
      x, Wq, Wk, Wv, wsq, wsk, wsvt);

  attn_kernel<<<BATCH * (SEQ_T / 32) * NUM_HEADS, 256, ATTN_SMEM_BYTES, stream>>>(
      wsq, wsk, wsvt, out);
}
